// BClassifier_19791209300131
// MI455X (gfx1250) — matrix operand order repaired and checked by reading, not compiled, not run on hardware
//
#include <hip/hip_runtime.h>

typedef __bf16 bf16_t;
typedef __attribute__((ext_vector_type(16))) bf16_t v16bf;
typedef __attribute__((ext_vector_type(8)))  float  v8f;

#define D 512
#define NNODE 256
#define KNB 8
#define NEDGE (NNODE * KNB)
#define NBAG 4
#define NROW 20000
#define NROWS_TOTAL (NBAG * NROW)
#define MPART_CHUNK 160
#define NCHUNKS 125   /* 125 * 160 == 20000 */

// ---------------------------------------------------------------------------
// Kernel W: convert aW1 (K x N, row-major) into transposed bf16 hi/lo [n][k]
// ---------------------------------------------------------------------------
__global__ void k_wconv(const float* __restrict__ W, bf16_t* __restrict__ hi,
                        bf16_t* __restrict__ lo) {
  int idx = blockIdx.x * blockDim.x + threadIdx.x;
  if (idx >= D * D) return;
  int n = idx / D, k = idx % D;
  float w = W[k * D + n];
  bf16_t h = (bf16_t)w;
  hi[n * D + k] = h;
  lo[n * D + k] = (bf16_t)(w - (float)h);
}

// ---------------------------------------------------------------------------
// Kernel A: a[row] = relu(x[row]@aW1 + ab1) @ aW2 + ab2 (fused, bf16-split WMMA)
// 4 waves per block, each wave owns one 16-row M-tile; all 512 N columns; K=512.
// B fragments for 8 N-tiles are staged in registers so 16 loads share one wait
// and the matrix pipe gets uninterrupted WMMA bursts.
// ---------------------------------------------------------------------------
__global__ void __launch_bounds__(128)
k_attn_gemm(const float* __restrict__ x, const bf16_t* __restrict__ whiT,
            const bf16_t* __restrict__ wloT, const float* __restrict__ ab1,
            const float* __restrict__ aW2, const float* __restrict__ ab2,
            float* __restrict__ aout) {
  const int lane = threadIdx.x & 31;
  const int wave = threadIdx.x >> 5;
  const int m15  = lane & 15;
  const int half = lane >> 4;
  const long rowbase = ((long)blockIdx.x * 4 + wave) * 16;
  const float* __restrict__ xrow = x + (rowbase + m15) * D;

  float rowacc[8];
#pragma unroll
  for (int v = 0; v < 8; ++v) rowacc[v] = 0.f;

  const v8f czero = {0.f, 0.f, 0.f, 0.f, 0.f, 0.f, 0.f, 0.f};

  for (int g = 0; g < 4; ++g) {
    v8f C[8];
#pragma unroll
    for (int t = 0; t < 8; ++t) C[t] = czero;

    for (int kc = 0; kc < 16; ++kc) {
      const int kb = kc * 32;
      // ---- A fragment: 16 fp32 values -> bf16 hi/lo (contiguous k = 16*half + e) ----
      float4 x0 = *(const float4*)(xrow + kb + half * 16);
      float4 x1 = *(const float4*)(xrow + kb + half * 16 + 4);
      float4 x2 = *(const float4*)(xrow + kb + half * 16 + 8);
      float4 x3 = *(const float4*)(xrow + kb + half * 16 + 12);
      float xs[16] = {x0.x, x0.y, x0.z, x0.w, x1.x, x1.y, x1.z, x1.w,
                      x2.x, x2.y, x2.z, x2.w, x3.x, x3.y, x3.z, x3.w};
      v16bf ahi, alo;
#pragma unroll
      for (int e = 0; e < 16; ++e) {
        float f = xs[e];
        bf16_t h = (bf16_t)f;
        ahi[e] = h;
        alo[e] = (bf16_t)(f - (float)h);
      }
      const int koffB = kb + half * 16;
      const bf16_t* __restrict__ bbase_hi = whiT + (g * 128 + m15) * D + koffB;
      const bf16_t* __restrict__ bbase_lo = wloT + (g * 128 + m15) * D + koffB;

      // ---- stage all 8 hi B-fragments (one clause, one wait) ----
      v16bf bfrag[8];
#pragma unroll
      for (int t = 0; t < 8; ++t)
        bfrag[t] = *(const v16bf*)(bbase_hi + t * 16 * D);
      // ---- 16 back-to-back WMMAs on the hi fragments ----
#pragma unroll
      for (int t = 0; t < 8; ++t) {
        C[t] = __builtin_amdgcn_wmma_f32_16x16x32_bf16(false, ahi, false, bfrag[t],
                                                       (short)0, C[t], false, false);
        C[t] = __builtin_amdgcn_wmma_f32_16x16x32_bf16(false, alo, false, bfrag[t],
                                                       (short)0, C[t], false, false);
      }
      // ---- stage all 8 lo B-fragments, then 8 correction WMMAs ----
#pragma unroll
      for (int t = 0; t < 8; ++t)
        bfrag[t] = *(const v16bf*)(bbase_lo + t * 16 * D);
#pragma unroll
      for (int t = 0; t < 8; ++t)
        C[t] = __builtin_amdgcn_wmma_f32_16x16x32_bf16(false, ahi, false, bfrag[t],
                                                       (short)0, C[t], false, false);
    }
    // ---- epilogue: +ab1, relu, * aW2, accumulate per-row ----
#pragma unroll
    for (int t = 0; t < 8; ++t) {
      const int n = g * 128 + t * 16 + m15;
      const float b1 = ab1[n];
      const float w2 = aW2[n];
#pragma unroll
      for (int v = 0; v < 8; ++v) {
        float h = C[t][v] + b1;
        h = fmaxf(h, 0.f);
        rowacc[v] += h * w2;
      }
    }
  }
  // reduce across the 16 lanes of each half (bits 0..3 only)
#pragma unroll
  for (int v = 0; v < 8; ++v) {
    float s = rowacc[v];
    s += __shfl_xor(s, 1, 32);
    s += __shfl_xor(s, 2, 32);
    s += __shfl_xor(s, 4, 32);
    s += __shfl_xor(s, 8, 32);
    rowacc[v] = s;
  }
  if (m15 == 0) {
    const float b2 = ab2[0];
#pragma unroll
    for (int v = 0; v < 8; ++v) aout[rowbase + half * 8 + v] = rowacc[v] + b2;
  }
}

// ---------------------------------------------------------------------------
// Softmax stats per bag: max and sum(exp(a-max))
// ---------------------------------------------------------------------------
__global__ void k_softmax_stats(const float* __restrict__ a, float* __restrict__ amax,
                                float* __restrict__ asum) {
  const int b = blockIdx.x;
  const float* ab = a + b * NROW;
  __shared__ float red[1024];
  const int t = threadIdx.x;
  float m = -3.402823466e38f;
  for (int i = t; i < NROW; i += 1024) m = fmaxf(m, ab[i]);
  red[t] = m;
  __syncthreads();
  for (int s = 512; s > 0; s >>= 1) {
    if (t < s) red[t] = fmaxf(red[t], red[t + s]);
    __syncthreads();
  }
  const float mx = red[0];
  __syncthreads();
  float sum = 0.f;
  for (int i = t; i < NROW; i += 1024) sum += expf(ab[i] - mx);
  red[t] = sum;
  __syncthreads();
  for (int s = 512; s > 0; s >>= 1) {
    if (t < s) red[t] += red[t + s];
    __syncthreads();
  }
  if (t == 0) {
    amax[b] = mx;
    asum[b] = red[0];
  }
}

// ---------------------------------------------------------------------------
// M partial sums: M[b] = sum_r alpha(b,r) * x[b,r,:]
// ---------------------------------------------------------------------------
__global__ void k_Mpart(const float* __restrict__ x, const float* __restrict__ a,
                        const float* __restrict__ amax, const float* __restrict__ asum,
                        float* __restrict__ Mpart) {
  const int chunk = blockIdx.x, b = blockIdx.y;
  __shared__ float al[MPART_CHUNK];
  const int t = threadIdx.x;  // 512
  if (t < MPART_CHUNK) {
    const int r = chunk * MPART_CHUNK + t;
    al[t] = (r < NROW) ? expf(a[b * NROW + r] - amax[b]) / asum[b] : 0.f;
  }
  __syncthreads();
  float acc = 0.f;
  const float* xb = x + (long)b * NROW * D;
  for (int i = 0; i < MPART_CHUNK; ++i) {
    const long r = chunk * MPART_CHUNK + i;
    acc += al[i] * xb[r * D + t];
  }
  Mpart[((b * NCHUNKS) + chunk) * D + t] = acc;
}

__global__ void k_Mreduce(const float* __restrict__ Mpart, float* __restrict__ M) {
  const int b = blockIdx.x, t = threadIdx.x;  // 512
  float s = 0.f;
  for (int c = 0; c < NCHUNKS; ++c) s += Mpart[((b * NCHUNKS) + c) * D + t];
  M[b * D + t] = s;
}

// ---------------------------------------------------------------------------
// logits_mlp = M @ bagW + bagB (4x4)
// ---------------------------------------------------------------------------
__global__ void k_logits_mlp(const float* __restrict__ M, const float* __restrict__ bagW,
                             const float* __restrict__ bagB, float* __restrict__ out) {
  const int idx = threadIdx.x;
  if (idx >= 16) return;
  const int r = idx >> 2, c = idx & 3;
  float s = bagB[c];
  for (int k = 0; k < D; ++k) s += M[r * D + k] * bagW[k * 4 + c];
  out[r * 4 + c] = s;
}

// ---------------------------------------------------------------------------
// x_concat = [M(4); rehearsal[0:252]]
// ---------------------------------------------------------------------------
__global__ void k_concat_xc(const float* __restrict__ M, const float* __restrict__ reh,
                            float* __restrict__ xc) {
  const int idx = blockIdx.x * blockDim.x + threadIdx.x;
  if (idx >= NNODE * D) return;
  const int r = idx >> 9;
  xc[idx] = (r < NBAG) ? M[idx] : reh[(r - NBAG) * D + (idx & (D - 1))];
}

// ---------------------------------------------------------------------------
// Generic small GEMM: out = act(A @ W + bias). act: 0 none, 1 relu, 2 lrelu(0.01)
// transA: A indexed [k*lda + row] instead of [row*lda + k]
// ---------------------------------------------------------------------------
__global__ void k_gemm(const float* __restrict__ A, const float* __restrict__ W,
                       const float* __restrict__ bias, float* __restrict__ out,
                       int M, int K, int N, int lda, int transA, int act) {
  const int col = blockIdx.x * blockDim.x + threadIdx.x;
  const int row = blockIdx.y * blockDim.y + threadIdx.y;
  if (col >= N || row >= M) return;
  float s = bias ? bias[col] : 0.f;
  if (transA) {
    for (int k = 0; k < K; ++k) s += A[k * lda + row] * W[k * N + col];
  } else {
    for (int k = 0; k < K; ++k) s += A[row * lda + k] * W[k * N + col];
  }
  if (act == 1) s = fmaxf(s, 0.f);
  else if (act == 2) s = (s > 0.f) ? s : 0.01f * s;
  out[row * N + col] = s;
}

// ---------------------------------------------------------------------------
// Row L2-normalize: xn = xf / max(||xf||, 1e-12)
// ---------------------------------------------------------------------------
__global__ void k_rownorm(const float* __restrict__ xf, float* __restrict__ xn) {
  const int r = blockIdx.x;
  const int t = threadIdx.x;  // 256
  __shared__ float red[256];
  const float v0 = xf[r * D + t];
  const float v1 = xf[r * D + 256 + t];
  red[t] = v0 * v0 + v1 * v1;
  __syncthreads();
  for (int s = 128; s > 0; s >>= 1) {
    if (t < s) red[t] += red[t + s];
    __syncthreads();
  }
  const float inv = 1.f / fmaxf(sqrtf(red[0]), 1e-12f);
  xn[r * D + t] = v0 * inv;
  xn[r * D + 256 + t] = v1 * inv;
}

__global__ void k_transpose(const float* __restrict__ in, float* __restrict__ out,
                            int R, int C) {
  const int idx = blockIdx.x * blockDim.x + threadIdx.x;
  if (idx >= R * C) return;
  const int r = idx / C, c = idx % C;
  out[c * R + r] = in[idx];
}

// ---------------------------------------------------------------------------
// top-k (K=8) per row of sim; ties -> lowest index (JAX semantics)
// ---------------------------------------------------------------------------
__global__ void k_topk(const float* __restrict__ sim, int* __restrict__ nbr) {
  const int r = blockIdx.x * blockDim.x + threadIdx.x;
  if (r >= NNODE) return;
  const float* row = sim + r * NNODE;
  int sel[KNB];
  for (int j = 0; j < KNB; ++j) {
    float best = -3.402823466e38f;
    int bi = 0;
    for (int c = 0; c < NNODE; ++c) {
      bool taken = false;
      for (int p = 0; p < j; ++p)
        if (sel[p] == c) taken = true;
      if (taken) continue;
      const float v = row[c];
      if (v > best) { best = v; bi = c; }
    }
    sel[j] = bi;
    nbr[r * KNB + j] = bi;
  }
}

__global__ void k_edge_attr(const float* __restrict__ xf, const int* __restrict__ nbr,
                            float* __restrict__ ea) {
  const int idx = blockIdx.x * blockDim.x + threadIdx.x;
  if (idx >= NNODE * D) return;
  const int n = idx >> 9, c = idx & (D - 1);
  float s = 0.f;
  for (int j = 0; j < KNB; ++j) s += xf[nbr[n * KNB + j] * D + c];
  ea[idx] = s * (1.f / KNB);
}

// ---------------------------------------------------------------------------
// Deterministic CSR build over e_src (single thread; 2048 edges)
// ---------------------------------------------------------------------------
__global__ void k_csr(const int* __restrict__ nbr, int* __restrict__ off,
                      int* __restrict__ list, float* __restrict__ dinv) {
  if (threadIdx.x != 0 || blockIdx.x != 0) return;
  __shared__ int cnt[NNODE];
  __shared__ int pos[NNODE];
  for (int n = 0; n < NNODE; ++n) cnt[n] = 0;
  for (int e = 0; e < NEDGE; ++e) cnt[nbr[e]]++;
  int run = 0;
  for (int n = 0; n < NNODE; ++n) {
    off[n] = run;
    pos[n] = run;
    run += cnt[n];
  }
  off[NNODE] = run;
  for (int e = 0; e < NEDGE; ++e) {
    const int s = nbr[e];
    list[pos[s]++] = e;
  }
  for (int n = 0; n < NNODE; ++n) dinv[n] = (cnt[n] > 0) ? 1.f / (float)cnt[n] : 0.f;
}

// ---------------------------------------------------------------------------
// Hypergraph-conv pieces
// ---------------------------------------------------------------------------
__global__ void k_pq(const float* __restrict__ xw, const float* __restrict__ hw,
                     const float* __restrict__ att, float* __restrict__ p,
                     float* __restrict__ q) {
  const int n = blockIdx.x * blockDim.x + threadIdx.x;
  if (n >= NNODE) return;
  float sp = 0.f, sq = 0.f;
  for (int k = 0; k < D; ++k) {
    sp += xw[n * D + k] * att[k];
    sq += hw[n * D + k] * att[D + k];
  }
  p[n] = sp;
  q[n] = sq;
}

__global__ void k_edge_a(const float* __restrict__ p, const float* __restrict__ q,
                         const int* __restrict__ nbr, float* __restrict__ a) {
  const int e = blockIdx.x * blockDim.x + threadIdx.x;
  if (e >= NEDGE) return;
  const float v = p[nbr[e]] + q[e >> 3];
  a[e] = (v > 0.f) ? v : 0.2f * v;
}

__global__ void k_seg_max(const float* __restrict__ a, const int* __restrict__ off,
                          const int* __restrict__ list, float* __restrict__ amax) {
  const int n = blockIdx.x * blockDim.x + threadIdx.x;
  if (n >= NNODE) return;
  float m = -3.402823466e38f;
  for (int i = off[n]; i < off[n + 1]; ++i) m = fmaxf(m, a[list[i]]);
  amax[n] = m;
}

__global__ void k_edge_exp(const float* __restrict__ a, const float* __restrict__ amax,
                           const int* __restrict__ nbr, float* __restrict__ ae) {
  const int e = blockIdx.x * blockDim.x + threadIdx.x;
  if (e >= NEDGE) return;
  ae[e] = expf(a[e] - amax[nbr[e]]);
}

__global__ void k_seg_sum(const float* __restrict__ ae, const int* __restrict__ off,
                          const int* __restrict__ list, float* __restrict__ asum) {
  const int n = blockIdx.x * blockDim.x + threadIdx.x;
  if (n >= NNODE) return;
  float s = 0.f;
  for (int i = off[n]; i < off[n + 1]; ++i) s += ae[list[i]];
  asum[n] = s;
}

__global__ void k_edge_alpha(const float* __restrict__ ae, const float* __restrict__ asum,
                             const int* __restrict__ nbr, float* __restrict__ alpha) {
  const int e = blockIdx.x * blockDim.x + threadIdx.x;
  if (e >= NEDGE) return;
  alpha[e] = ae[e] / asum[nbr[e]];
}

__global__ void k_edge_feat(const float* __restrict__ xw, const int* __restrict__ nbr,
                            const float* __restrict__ alpha, float* __restrict__ ef) {
  const int idx = blockIdx.x * blockDim.x + threadIdx.x;
  if (idx >= NNODE * D) return;
  const int m = idx >> 9, c = idx & (D - 1);
  float s = 0.f;
  for (int j = 0; j < KNB; ++j) {
    const int e = m * KNB + j;
    s += alpha[e] * xw[nbr[e] * D + c];
  }
  ef[idx] = s * (1.f / KNB);  // Binv == 1/8 (each dst has exactly K edges)
}

__global__ void k_conv_out(const float* __restrict__ ef, const int* __restrict__ off,
                           const int* __restrict__ list, const float* __restrict__ alpha,
                           const float* __restrict__ dinv, const float* __restrict__ bias,
                           float* __restrict__ out) {
  const int idx = blockIdx.x * blockDim.x + threadIdx.x;
  if (idx >= NNODE * D) return;
  const int n = idx >> 9, c = idx & (D - 1);
  float s = 0.f;
  for (int i = off[n]; i < off[n + 1]; ++i) {
    const int e = list[i];
    s += alpha[e] * ef[(e >> 3) * D + c];
  }
  out[idx] = dinv[n] * s + bias[c];
}

// ---------------------------------------------------------------------------
// graph_norm (per-column over 256 nodes) + lrelu(0.01)
// ---------------------------------------------------------------------------
__global__ void k_gnorm(const float* __restrict__ g, const float* __restrict__ w,
                        const float* __restrict__ b, const float* __restrict__ ms,
                        float* __restrict__ out) {
  const int c = blockIdx.x;   // 512 columns
  const int t = threadIdx.x;  // 256 nodes
  __shared__ float red[NNODE];
  const float v = g[t * D + c];
  red[t] = v;
  __syncthreads();
  for (int s = 128; s > 0; s >>= 1) {
    if (t < s) red[t] += red[t + s];
    __syncthreads();
  }
  const float mean = red[0] * (1.f / NNODE);
  __syncthreads();
  const float o = v - mean * ms[c];
  red[t] = o * o;
  __syncthreads();
  for (int s = 128; s > 0; s >>= 1) {
    if (t < s) red[t] += red[t + s];
    __syncthreads();
  }
  const float var = red[0] * (1.f / NNODE);
  float y = w[c] * o / sqrtf(var + 1e-5f) + b[c];
  out[t * D + c] = (y > 0.f) ? y : 0.01f * y;
}

__global__ void k_outcat(const float* __restrict__ xf, const float* __restrict__ o1,
                         const float* __restrict__ o2, float* __restrict__ oc) {
  const int idx = blockIdx.x * blockDim.x + threadIdx.x;
  if (idx >= NNODE * 1024) return;
  const int r = idx >> 10, c = idx & 1023;
  float v;
  if (c < 512) v = xf[r * D + c];
  else if (c < 768) v = o1[r * 256 + (c - 512)];
  else v = o2[r * 256 + (c - 768)];
  oc[idx] = v;
}

__global__ void k_sigmoid_center(const float* __restrict__ S, float* __restrict__ s) {
  __shared__ float red[1024];
  const int t = threadIdx.x;
  const float v = 1.f / (1.f + expf(-S[t]));
  red[t] = v;
  __syncthreads();
  for (int sh = 512; sh > 0; sh >>= 1) {
    if (t < sh) red[t] += red[t + sh];
    __syncthreads();
  }
  s[t] = v - red[0] * (1.f / 1024.f);
}

__global__ void k_logits_graph(const float* __restrict__ oc, const float* __restrict__ s,
                               const float* __restrict__ cW, const float* __restrict__ cb,
                               float* __restrict__ dst) {
  const int idx = threadIdx.x;
  if (idx >= 16) return;
  const int r = idx >> 2, c = idx & 3;
  float acc = cb[c];
  for (int k = 0; k < 1024; ++k) acc += oc[r * 1024 + k] * s[k] * cW[k * 4 + c];
  dst[r * 4 + c] = acc;
}

// ---------------------------------------------------------------------------
extern "C" void kernel_launch(void* const* d_in, const int* in_sizes, int n_in,
                              void* d_out, int out_size, void* d_ws, size_t ws_size,
                              hipStream_t stream) {
  (void)in_sizes; (void)n_in; (void)out_size; (void)ws_size;
  const float* x    = (const float*)d_in[0];
  const float* aW1  = (const float*)d_in[1];
  const float* ab1  = (const float*)d_in[2];
  const float* aW2  = (const float*)d_in[3];
  const float* ab2  = (const float*)d_in[4];
  const float* bagW = (const float*)d_in[5];
  const float* bagB = (const float*)d_in[6];
  const float* reh  = (const float*)d_in[7];
  const float* dW1  = (const float*)d_in[8];
  const float* db1  = (const float*)d_in[9];
  const float* dW2  = (const float*)d_in[10];
  const float* db2  = (const float*)d_in[11];
  const float* h1W  = (const float*)d_in[12];
  const float* h1att= (const float*)d_in[13];
  const float* h1b  = (const float*)d_in[14];
  const float* h2W  = (const float*)d_in[15];
  const float* h2att= (const float*)d_in[16];
  const float* h2b  = (const float*)d_in[17];
  const float* n1w  = (const float*)d_in[18];
  const float* n1b  = (const float*)d_in[19];
  const float* n1ms = (const float*)d_in[20];
  const float* n2w  = (const float*)d_in[21];
  const float* n2b  = (const float*)d_in[22];
  const float* n2ms = (const float*)d_in[23];
  const float* f1W  = (const float*)d_in[24];
  const float* f1b  = (const float*)d_in[25];
  const float* f2W  = (const float*)d_in[26];
  const float* f2b  = (const float*)d_in[27];
  const float* gW1  = (const float*)d_in[28];
  const float* gb1  = (const float*)d_in[29];
  const float* gW2  = (const float*)d_in[30];
  const float* gb2  = (const float*)d_in[31];
  const float* cW   = (const float*)d_in[32];
  const float* cb   = (const float*)d_in[33];
  float* out = (float*)d_out;
  char* ws = (char*)d_ws;

  size_t off = 0;
  auto alloc = [&](size_t bytes) -> char* {
    char* p = ws + off;
    off = (off + bytes + 255) & ~(size_t)255;
    return p;
  };

  bf16_t* whiT  = (bf16_t*)alloc(D * D * sizeof(bf16_t));
  bf16_t* wloT  = (bf16_t*)alloc(D * D * sizeof(bf16_t));
  float* abuf   = (float*)alloc(NROWS_TOTAL * sizeof(float));
  float* amax4  = (float*)alloc(NBAG * sizeof(float));
  float* asum4  = (float*)alloc(NBAG * sizeof(float));
  float* Mpart  = (float*)alloc((size_t)NBAG * NCHUNKS * D * sizeof(float));
  float* Mbuf   = (float*)alloc(NBAG * D * sizeof(float));
  float* xc     = (float*)alloc(NNODE * D * sizeof(float));
  float* t1     = (float*)alloc(NNODE * 256 * sizeof(float));
  float* xf     = (float*)alloc(NNODE * D * sizeof(float));
  float* xn     = (float*)alloc(NNODE * D * sizeof(float));
  float* xnT    = (float*)alloc(D * NNODE * sizeof(float));
  float* sim    = (float*)alloc(NNODE * NNODE * sizeof(float));
  int*   nbr    = (int*)alloc(NEDGE * sizeof(int));
  int*   coff   = (int*)alloc((NNODE + 1) * sizeof(int));
  int*   clist  = (int*)alloc(NEDGE * sizeof(int));
  float* dinv   = (float*)alloc(NNODE * sizeof(float));
  float* ea     = (float*)alloc(NNODE * D * sizeof(float));
  float* xw     = (float*)alloc(NNODE * D * sizeof(float));
  float* hw     = (float*)alloc(NNODE * D * sizeof(float));
  float* pbuf   = (float*)alloc(NNODE * sizeof(float));
  float* qbuf   = (float*)alloc(NNODE * sizeof(float));
  float* eabuf  = (float*)alloc(NEDGE * sizeof(float));
  float* aebuf  = (float*)alloc(NEDGE * sizeof(float));
  float* alphab = (float*)alloc(NEDGE * sizeof(float));
  float* amaxn  = (float*)alloc(NNODE * sizeof(float));
  float* asumn  = (float*)alloc(NNODE * sizeof(float));
  float* ef     = (float*)alloc(NNODE * D * sizeof(float));
  float* g1     = (float*)alloc(NNODE * D * sizeof(float));
  float* gact   = (float*)alloc(NNODE * D * sizeof(float));
  float* out1   = (float*)alloc(NNODE * 256 * sizeof(float));
  float* out2   = (float*)alloc(NNODE * 256 * sizeof(float));
  float* ocat   = (float*)alloc(NNODE * 1024 * sizeof(float));
  float* T1     = (float*)alloc(1024 * 256 * sizeof(float));
  float* Sbuf   = (float*)alloc(1024 * sizeof(float));
  float* svec   = (float*)alloc(1024 * sizeof(float));

  const dim3 gblk(64, 4);
  auto ggrid = [](int N, int M) { return dim3((N + 63) / 64, (M + 3) / 4); };

  // ---- Stage 1: attention MLP (dominant GEMM) ----
  k_wconv<<<(D * D + 255) / 256, 256, 0, stream>>>(aW1, whiT, wloT);
  k_attn_gemm<<<NROWS_TOTAL / 64, 128, 0, stream>>>(x, whiT, wloT, ab1, aW2, ab2, abuf);
  k_softmax_stats<<<NBAG, 1024, 0, stream>>>(abuf, amax4, asum4);
  k_Mpart<<<dim3(NCHUNKS, NBAG), D, 0, stream>>>(x, abuf, amax4, asum4, Mpart);
  k_Mreduce<<<NBAG, D, 0, stream>>>(Mpart, Mbuf);
  k_logits_mlp<<<1, 32, 0, stream>>>(Mbuf, bagW, bagB, out);  // out[0..15]

  // ---- Stage 2: node features + graph construction ----
  k_concat_xc<<<(NNODE * D + 255) / 256, 256, 0, stream>>>(Mbuf, reh, xc);
  k_gemm<<<ggrid(256, NNODE), gblk, 0, stream>>>(xc, dW1, db1, t1, NNODE, D, 256, D, 0, 2);
  k_gemm<<<ggrid(D, NNODE), gblk, 0, stream>>>(t1, dW2, db2, xf, NNODE, 256, D, 256, 0, 2);
  k_rownorm<<<NNODE, 256, 0, stream>>>(xf, xn);
  k_transpose<<<(NNODE * D + 255) / 256, 256, 0, stream>>>(xn, xnT, NNODE, D);
  k_gemm<<<ggrid(NNODE, NNODE), gblk, 0, stream>>>(xn, xnT, nullptr, sim, NNODE, D, NNODE, D, 0, 0);
  k_topk<<<1, NNODE, 0, stream>>>(sim, nbr);
  k_csr<<<1, 32, 0, stream>>>(nbr, coff, clist, dinv);
  k_edge_attr<<<(NNODE * D + 255) / 256, 256, 0, stream>>>(xf, nbr, ea);

  // ---- Hypergraph conv (shared lambda) ----
  auto run_conv = [&](const float* xin, const float* W, const float* att,
                      const float* bias, float* gout) {
    k_gemm<<<ggrid(D, NNODE), gblk, 0, stream>>>(xin, W, nullptr, xw, NNODE, D, D, D, 0, 0);
    k_gemm<<<ggrid(D, NNODE), gblk, 0, stream>>>(ea, W, nullptr, hw, NNODE, D, D, D, 0, 0);
    k_pq<<<1, NNODE, 0, stream>>>(xw, hw, att, pbuf, qbuf);
    k_edge_a<<<NEDGE / 256, 256, 0, stream>>>(pbuf, qbuf, nbr, eabuf);
    k_seg_max<<<1, NNODE, 0, stream>>>(eabuf, coff, clist, amaxn);
    k_edge_exp<<<NEDGE / 256, 256, 0, stream>>>(eabuf, amaxn, nbr, aebuf);
    k_seg_sum<<<1, NNODE, 0, stream>>>(aebuf, coff, clist, asumn);
    k_edge_alpha<<<NEDGE / 256, 256, 0, stream>>>(aebuf, asumn, nbr, alphab);
    k_edge_feat<<<(NNODE * D + 255) / 256, 256, 0, stream>>>(xw, nbr, alphab, ef);
    k_conv_out<<<(NNODE * D + 255) / 256, 256, 0, stream>>>(ef, coff, clist, alphab, dinv,
                                                            bias, gout);
  };

  // conv1 + norm1 + fc1
  run_conv(xf, h1W, h1att, h1b, g1);
  k_gnorm<<<D, NNODE, 0, stream>>>(g1, n1w, n1b, n1ms, gact);
  k_gemm<<<ggrid(256, NNODE), gblk, 0, stream>>>(gact, f1W, f1b, out1, NNODE, D, 256, D, 0, 2);

  // conv2 + norm2 + fc2 (conv2 consumes gact = lrelu(norm1(g1)))
  run_conv(gact, h2W, h2att, h2b, g1);
  k_gnorm<<<D, NNODE, 0, stream>>>(g1, n2w, n2b, n2ms, gact);
  k_gemm<<<ggrid(256, NNODE), gblk, 0, stream>>>(gact, f2W, f2b, out2, NNODE, D, 256, D, 0, 2);

  // ---- Stage 3: column scoring + final logits ----
  k_outcat<<<(NNODE * 1024 + 255) / 256, 256, 0, stream>>>(xf, out1, out2, ocat);
  k_gemm<<<ggrid(256, 1024), gblk, 0, stream>>>(ocat, gW1, gb1, T1, 1024, 256, 256, 1024, 1, 1);
  k_gemm<<<ggrid(1, 1024), gblk, 0, stream>>>(T1, gW2, gb2, Sbuf, 1024, 256, 1, 256, 0, 0);
  k_sigmoid_center<<<1, 1024, 0, stream>>>(Sbuf, svec);
  k_logits_graph<<<1, 32, 0, stream>>>(ocat, svec, cW, cb, out + 16);  // out[16..31]
}
